// MemoryRamModule_batch_25623774888599
// MI455X (gfx1250) — matrix operand order repaired and checked by reading, not compiled, not run on hardware
//
#include <hip/hip_runtime.h>
#include <stdint.h>

typedef __bf16 bf16_t;
typedef __attribute__((ext_vector_type(16))) __bf16 v16bf;
typedef __attribute__((ext_vector_type(8)))  __bf16 v8bf;
typedef __attribute__((ext_vector_type(8)))  float  v8f;

#define B_SZ   128
#define T_SZ   256
#define DIN    1024
#define DH     512
#define MBANK  100
#define MPAD   112
#define KXH    (DIN + DH)                 // 1536
#define NALL   (MPAD + MPAD + DH + DH)    // 1248 = 78 n-tiles = 39 dual-tiles
#define OFF_RP 0
#define OFF_WP MPAD                       // 112
#define OFF_C  (2*MPAD)                   // 224
#define OFF_H  (2*MPAD + DH)              // 736
#define NWT1   39                         // dual n-tiles in the fused GEMM

__device__ __forceinline__ bf16_t f2bf(float f) {
  union { float f; uint32_t u; } x; x.f = f;
  uint32_t u = x.u;
  uint32_t r = (u + 0x7FFFu + ((u >> 16) & 1u)) >> 16;   // round-to-nearest-even
  union { uint16_t s; bf16_t b; } y; y.s = (uint16_t)r;
  return y.b;
}

// ---------------- init kernels (run once per launch) ----------------

__global__ void k_init_wtall(const float* __restrict__ W_rp, const float* __restrict__ W_wp,
                             const float* __restrict__ W_c,  const float* __restrict__ Wxh,
                             const float* __restrict__ Whh,  bf16_t* __restrict__ WT) {
  int idx = blockIdx.x * 256 + threadIdx.x;
  if (idx >= NALL * KXH) return;
  int n = idx / KXH, k = idx % KXH;
  float v = 0.f;
  if (n < MPAD)            { if (n < MBANK) v = W_rp[k * MBANK + n]; }
  else if (n < 2 * MPAD)   { int nn = n - MPAD; if (nn < MBANK) v = W_wp[k * MBANK + nn]; }
  else if (n < OFF_H)      { int nn = n - OFF_C; v = W_c[k * DH + nn]; }
  else                     { int nn = n - OFF_H;
                             v = (k < DIN) ? Wxh[k * DH + nn] : Whh[(k - DIN) * DH + nn]; }
  WT[idx] = f2bf(v);
}

__global__ void k_init_rh_bias(const float* __restrict__ Wrh,
                               const float* __restrict__ b_rp, const float* __restrict__ b_wp,
                               const float* __restrict__ b_c,  const float* __restrict__ bh,
                               bf16_t* __restrict__ WT_rh, float* __restrict__ bias) {
  int idx = blockIdx.x * 256 + threadIdx.x;
  if (idx < DH * DH) {
    int n = idx / DH, k = idx % DH;
    WT_rh[idx] = f2bf(Wrh[k * DH + n]);
  } else if (idx < DH * DH + NALL) {
    int j = idx - DH * DH;
    float v = 0.f;
    if (j < MPAD)          { if (j < MBANK) v = b_rp[j]; }
    else if (j < 2 * MPAD) { int n = j - MPAD; if (n < MBANK) v = b_wp[n]; }
    else if (j < OFF_H)    v = b_c[j - OFF_C];
    else                   v = bh[j - OFF_H];
    bias[j] = v;
  }
}

__global__ void k_init_state(bf16_t* __restrict__ h_st, float* __restrict__ mem) {
  int idx = blockIdx.x * 256 + threadIdx.x;
  if (idx < B_SZ * MBANK * DH) mem[idx] = 0.f;
  if (idx < B_SZ * DH)         h_st[idx] = f2bf(0.f);
}

// ---------------- per-step kernel 1: fused GEMM (128 x 1536 x 1248) ----------------
// out1[b][n] = xh[b] . WT_all[n] + bias[n]   (relu applied on the c-slice only)
// 8 waves/block; each wave owns TWO 16x16 n-tiles (independent accumulators -> WMMA ILP)
#define AST1 (KXH + 8)   // LDS row stride in halves (bank-conflict pad)

__global__ __launch_bounds__(256) void k_gemm1(const float*  __restrict__ x_frames,
                                               const bf16_t* __restrict__ h_st,
                                               const bf16_t* __restrict__ WT,
                                               const float*  __restrict__ bias,
                                               float* __restrict__ out1, int t) {
  __shared__ bf16_t alds[16 * AST1];
  const int bt  = blockIdx.x;          // batch tile (16 rows)
  const int tid = threadIdx.x;

  // stage A tile xh (16 x 1536) in bf16: k<1024 from x_t (fp32->bf16), else from h state
  for (int i = tid; i < 16 * KXH; i += 256) {
    int m = i / KXH, k = i % KXH;
    int row = bt * 16 + m;
    if (k < DIN) alds[m * AST1 + k] = f2bf(x_frames[((size_t)row * T_SZ + t) * DIN + k]);
    else         alds[m * AST1 + k] = h_st[row * DH + (k - DIN)];
  }
  __syncthreads();

  const int wave = tid >> 5, lane = tid & 31;
  const int wt = blockIdx.y * 8 + wave;         // dual-tile index 0..38
  if (wt >= NWT1) return;                       // wave-uniform -> EXEC stays all-ones
  const int n0 = wt * 32;
  const int ml = lane & 15, hl = lane >> 4;

  const bf16_t* __restrict__ brow0 = WT + (size_t)(n0 + ml) * KXH + hl * 16;
  const bf16_t* __restrict__ brow1 = brow0 + (size_t)16 * KXH;
  const bf16_t* __restrict__ arow  = alds + ml * AST1 + hl * 16;

  v8f acc0 = {}, acc1 = {};
  #pragma unroll 2
  for (int kb = 0; kb < KXH; kb += 32) {
    v8bf a_lo = *(const v8bf*)(arow + kb);        // k = kb + 16*hl .. +7
    v8bf a_hi = *(const v8bf*)(arow + kb + 8);    // k = kb + 16*hl + 8 .. +15
    v16bf a;
    #pragma unroll
    for (int e = 0; e < 8; ++e) { a[e] = a_lo[e]; a[e + 8] = a_hi[e]; }
    v16bf b0 = *(const v16bf*)(brow0 + kb);       // k = kb + 16*hl .. +15 (contig 32B)
    v16bf b1 = *(const v16bf*)(brow1 + kb);
    acc0 = __builtin_amdgcn_wmma_f32_16x16x32_bf16(false, a, false, b0, (short)0, acc0,
                                                   false, false);
    acc1 = __builtin_amdgcn_wmma_f32_16x16x32_bf16(false, a, false, b1, (short)0, acc1,
                                                   false, false);
  }

  #pragma unroll
  for (int half_tile = 0; half_tile < 2; ++half_tile) {
    const int n = n0 + half_tile * 16 + ml;
    const float bs = bias[n];
    const bool do_relu = (n >= OFF_C) && (n < OFF_H);
    const v8f& acc = half_tile ? acc1 : acc0;
    #pragma unroll
    for (int r = 0; r < 8; ++r) {
      int m = r + 8 * hl;                         // C layout: VGPR r, lane half -> M
      float v = acc[r] + bs;
      if (do_relu) v = fmaxf(v, 0.f);
      out1[(size_t)(bt * 16 + m) * NALL + n] = v;
    }
  }
}

// ---------------- per-step kernel 2: softmax + memory read + gated blend ----------------
__global__ __launch_bounds__(256) void k_read_update(const float* __restrict__ out1,
                                                     float* __restrict__ mem,
                                                     bf16_t* __restrict__ r_bf) {
  const int b = blockIdx.x;
  const int tid = threadIdx.x;
  __shared__ float s_ar[MBANK];
  __shared__ float s_aw[MBANK];
  __shared__ float s_c[DH];
  __shared__ float s_inv[2];

  const float* __restrict__ lg = out1 + (size_t)b * NALL;
  if (tid < MBANK)                            s_ar[tid]       = lg[OFF_RP + tid];
  else if (tid >= 128 && tid < 128 + MBANK)   s_aw[tid - 128] = lg[OFF_WP + (tid - 128)];
  for (int i = tid; i < DH; i += 256)         s_c[i]          = lg[OFF_C + i];
  __syncthreads();

  if (tid == 0) {                // rp softmax (serial over 100: negligible)
    float mx = -1e30f;
    for (int i = 0; i < MBANK; ++i) mx = fmaxf(mx, s_ar[i]);
    float sm = 0.f;
    for (int i = 0; i < MBANK; ++i) { float e = __expf(s_ar[i] - mx); s_ar[i] = e; sm += e; }
    s_inv[0] = 1.f / sm;
  } else if (tid == 32) {        // wp softmax in a parallel wave
    float mx = -1e30f;
    for (int i = 0; i < MBANK; ++i) mx = fmaxf(mx, s_aw[i]);
    float sm = 0.f;
    for (int i = 0; i < MBANK; ++i) { float e = __expf(s_aw[i] - mx); s_aw[i] = e; sm += e; }
    s_inv[1] = 1.f / sm;
  }
  __syncthreads();
  if (tid < MBANK)                          s_ar[tid]       *= s_inv[0];
  else if (tid >= 128 && tid < 128 + MBANK) s_aw[tid - 128] *= s_inv[1];
  __syncthreads();

  // r[b][h] = sum_m ar[m] * mem[b][m][h]   (reads OLD mem; L2-resident)
  float* __restrict__ memb = mem + (size_t)b * MBANK * DH;
  for (int h = tid; h < DH; h += 256) {
    float acc = 0.f;
    #pragma unroll 4
    for (int m = 0; m < MBANK; ++m) acc += s_ar[m] * memb[m * DH + h];
    r_bf[(size_t)b * DH + h] = f2bf(acc);
  }
  __syncthreads();   // all reads of old mem complete before the gated overwrite

  for (int i = tid; i < MBANK * DH; i += 256) {
    int m = i / DH, h = i % DH;
    float a = s_aw[m];
    memb[i] = a * s_c[h] + (1.f - a) * memb[i];
  }
}

// ---------------- per-step kernel 3: h_new = relu(hpre + r @ Wrh) ----------------
// 4 waves/block; each wave owns TWO 16x16 n-tiles (32 n-tiles -> 16 dual-tiles, grid.y=4)
#define AST3 (DH + 8)

__global__ __launch_bounds__(128) void k_gemm_h(const bf16_t* __restrict__ r_bf,
                                                const bf16_t* __restrict__ WT_rh,
                                                const float*  __restrict__ out1,
                                                float* __restrict__ out,
                                                bf16_t* __restrict__ h_st, int t) {
  __shared__ bf16_t alds[16 * AST3];
  const int bt  = blockIdx.x;
  const int tid = threadIdx.x;
  for (int i = tid; i < 16 * DH; i += 128) {
    int m = i / DH, k = i % DH;
    alds[m * AST3 + k] = r_bf[(size_t)(bt * 16 + m) * DH + k];
  }
  __syncthreads();

  const int wave = tid >> 5, lane = tid & 31;
  const int wt = blockIdx.y * 4 + wave;              // dual-tile 0..15
  const int n0 = wt * 32;
  const int ml = lane & 15, hl = lane >> 4;
  const bf16_t* __restrict__ brow0 = WT_rh + (size_t)(n0 + ml) * DH + hl * 16;
  const bf16_t* __restrict__ brow1 = brow0 + (size_t)16 * DH;
  const bf16_t* __restrict__ arow  = alds + ml * AST3 + hl * 16;

  v8f acc0 = {}, acc1 = {};
  #pragma unroll 2
  for (int kb = 0; kb < DH; kb += 32) {
    v8bf a_lo = *(const v8bf*)(arow + kb);
    v8bf a_hi = *(const v8bf*)(arow + kb + 8);
    v16bf a;
    #pragma unroll
    for (int e = 0; e < 8; ++e) { a[e] = a_lo[e]; a[e + 8] = a_hi[e]; }
    v16bf b0 = *(const v16bf*)(brow0 + kb);
    v16bf b1 = *(const v16bf*)(brow1 + kb);
    acc0 = __builtin_amdgcn_wmma_f32_16x16x32_bf16(false, a, false, b0, (short)0, acc0,
                                                   false, false);
    acc1 = __builtin_amdgcn_wmma_f32_16x16x32_bf16(false, a, false, b1, (short)0, acc1,
                                                   false, false);
  }

  #pragma unroll
  for (int half_tile = 0; half_tile < 2; ++half_tile) {
    const int n = n0 + half_tile * 16 + ml;
    const v8f& acc = half_tile ? acc1 : acc0;
    #pragma unroll
    for (int r = 0; r < 8; ++r) {
      int m = r + 8 * hl;
      int row = bt * 16 + m;
      float v = acc[r] + out1[(size_t)row * NALL + OFF_H + n];
      v = fmaxf(v, 0.f);
      out[((size_t)row * T_SZ + t) * DH + n] = v;     // hs output (B,T,H)
      h_st[(size_t)row * DH + n] = f2bf(v);           // recurrent state for t+1
    }
  }
}

// ---------------- launch ----------------
extern "C" void kernel_launch(void* const* d_in, const int* in_sizes, int n_in,
                              void* d_out, int out_size, void* d_ws, size_t ws_size,
                              hipStream_t stream) {
  const float* x    = (const float*)d_in[0];
  const float* W_c  = (const float*)d_in[1];
  const float* b_c  = (const float*)d_in[2];
  const float* W_rp = (const float*)d_in[3];
  const float* b_rp = (const float*)d_in[4];
  const float* W_wp = (const float*)d_in[5];
  const float* b_wp = (const float*)d_in[6];
  const float* Wxh  = (const float*)d_in[7];
  const float* Wrh  = (const float*)d_in[8];
  const float* Whh  = (const float*)d_in[9];
  const float* bh   = (const float*)d_in[10];
  // d_in[11] = nImg (== T_SZ == 256 per setup_inputs)

  char* ws = (char*)d_ws;
  size_t off = 0;
  auto alloc = [&](size_t bytes) -> void* {
    void* p = ws + off;
    off = (off + bytes + 255) & ~(size_t)255;
    return p;
  };
  bf16_t* WT_all = (bf16_t*)alloc((size_t)NALL * KXH * 2);     // 3.84 MB
  bf16_t* WT_rh  = (bf16_t*)alloc((size_t)DH * DH * 2);        // 0.5 MB
  float*  bias   = (float*) alloc((size_t)NALL * 4);
  bf16_t* h_st   = (bf16_t*)alloc((size_t)B_SZ * DH * 2);
  bf16_t* r_bf   = (bf16_t*)alloc((size_t)B_SZ * DH * 2);
  float*  out1   = (float*) alloc((size_t)B_SZ * NALL * 4);
  float*  mem    = (float*) alloc((size_t)B_SZ * MBANK * DH * 4);  // 26.2 MB (L2-resident)
  (void)in_sizes; (void)n_in; (void)out_size; (void)ws_size;

  {
    int tot = NALL * KXH;
    k_init_wtall<<<(tot + 255) / 256, 256, 0, stream>>>(W_rp, W_wp, W_c, Wxh, Whh, WT_all);
    int tot2 = DH * DH + NALL;
    k_init_rh_bias<<<(tot2 + 255) / 256, 256, 0, stream>>>(Wrh, b_rp, b_wp, b_c, bh, WT_rh, bias);
    int tot3 = B_SZ * MBANK * DH;
    k_init_state<<<(tot3 + 255) / 256, 256, 0, stream>>>(h_st, mem);
  }

  float* outp = (float*)d_out;
  for (int t = 0; t < T_SZ; ++t) {
    k_gemm1<<<dim3(B_SZ / 16, 5), 256, 0, stream>>>(x, h_st, WT_all, bias, out1, t);
    k_read_update<<<B_SZ, 256, 0, stream>>>(out1, mem, r_bf);
    k_gemm_h<<<dim3(B_SZ / 16, 4), 128, 0, stream>>>(r_bf, WT_rh, out1, outp, h_st, t);
  }
}
